// PointnetEquivFPModule_38044820308016
// MI455X (gfx1250) — matrix operand order repaired and checked by reading, not compiled, not run on hardware
//
#include <hip/hip_runtime.h>

// ---------------------------------------------------------------------------
// PointNet FP module on MI455X (gfx1250, wave32, WMMA):
//   knn -> interp/concat (bf16 pack) -> GEMM1 (wmma bf16, async-LDS B tiles)
//   -> BN stats -> BN+ReLU -> GEMM2 -> BN stats -> BN+ReLU -> out f32
// ---------------------------------------------------------------------------

typedef __attribute__((ext_vector_type(16))) __bf16 v16bf;
typedef __attribute__((ext_vector_type(8)))  float  v8f;

#define BB    8
#define NPTS  2048
#define MPTS  512
#define CCH   256
#define NRR   4
#define COLS  65536        // B * NR * N
#define K1DIM 512
#define K2DIM 256

#define BROW_STRIDE 144            // 128B row + 16B pad (bank-conflict-free)
#define BTILE_BYTES (32 * BROW_STRIDE)   // 4608 per buffer

static __device__ __forceinline__ unsigned short f2bf(float f) {
  unsigned int u = __float_as_uint(f);
  unsigned int r = u + 0x7FFFu + ((u >> 16) & 1u);   // round-to-nearest-even
  return (unsigned short)(r >> 16);
}

// --------------------------- 1) 3-NN + weights -----------------------------
__global__ void knn_kernel(const float* __restrict__ unknown,
                           const float* __restrict__ known,
                           int* __restrict__ idx, float* __restrict__ wgt) {
  __shared__ float kx[MPTS], ky[MPTS], kz[MPTS];
  const int b = blockIdx.y;
  const int n = blockIdx.x * 256 + threadIdx.x;
  const float* kb = known + (size_t)b * MPTS * 3;
  for (int m = threadIdx.x; m < MPTS; m += 256) {
    kx[m] = kb[m * 3 + 0];
    ky[m] = kb[m * 3 + 1];
    kz[m] = kb[m * 3 + 2];
  }
  __syncthreads();
  const float* up = unknown + ((size_t)b * NPTS + n) * 3;
  const float ux = up[0], uy = up[1], uz = up[2];
  float d0 = 3.4e38f, d1 = 3.4e38f, d2 = 3.4e38f;
  int   i0 = 0, i1 = 0, i2 = 0;
  for (int m = 0; m < MPTS; ++m) {
    const float dx = ux - kx[m], dy = uy - ky[m], dz = uz - kz[m];
    const float d = dx * dx + dy * dy + dz * dz;
    if (d < d0)      { d2 = d1; i2 = i1; d1 = d0; i1 = i0; d0 = d; i0 = m; }
    else if (d < d1) { d2 = d1; i2 = i1; d1 = d;  i1 = m; }
    else if (d < d2) { d2 = d;  i2 = m; }
  }
  d0 = fmaxf(d0, 0.f); d1 = fmaxf(d1, 0.f); d2 = fmaxf(d2, 0.f);
  const float r0 = 1.f / (sqrtf(d0) + 1e-8f);
  const float r1 = 1.f / (sqrtf(d1) + 1e-8f);
  const float r2 = 1.f / (sqrtf(d2) + 1e-8f);
  const float rs = 1.f / (r0 + r1 + r2);
  const size_t o = ((size_t)b * NPTS + n) * 3;
  idx[o + 0] = i0; idx[o + 1] = i1; idx[o + 2] = i2;
  wgt[o + 0] = r0 * rs; wgt[o + 1] = r1 * rs; wgt[o + 2] = r2 * rs;
}

// --------------------------- 2) pack weights to bf16 -----------------------
__global__ void pack_w(const float* __restrict__ W1, const float* __restrict__ W2,
                       unsigned short* __restrict__ W1p, unsigned short* __restrict__ W2p) {
  const int i = blockIdx.x * 256 + threadIdx.x;
  if (i < CCH * K1DIM) W1p[i] = f2bf(W1[i]);
  if (i < CCH * K2DIM) W2p[i] = f2bf(W2[i]);
}

// ------------- 3) gather + interpolate + concat -> X bf16 [512][65536] -----
__global__ void build_x(const float* __restrict__ kf, const float* __restrict__ uf,
                        const int* __restrict__ idx, const float* __restrict__ wgt,
                        unsigned short* __restrict__ X) {
  const int col = blockIdx.x * 256 + threadIdx.x;     // 0..65535  (b,r,n)
  const int k   = blockIdx.y;                         // 0..511
  const int b   = col >> 13;
  const int rem = col & 8191;
  const int r   = rem >> 11;
  const int n   = rem & 2047;
  float v;
  if (k < CCH) {                                      // interpolated known feats
    const size_t o = ((size_t)b * NPTS + n) * 3;
    const int   j0 = idx[o + 0], j1 = idx[o + 1], j2 = idx[o + 2];
    const float w0 = wgt[o + 0], w1 = wgt[o + 1], w2 = wgt[o + 2];
    const float* f = kf + (size_t)b * (CCH * NRR * MPTS) + (size_t)k * (NRR * MPTS) + (size_t)r * MPTS;
    v = w0 * f[j0] + w1 * f[j1] + w2 * f[j2];
  } else {                                            // skip connection
    v = uf[(size_t)b * (CCH * NRR * NPTS) + (size_t)(k - CCH) * (NRR * NPTS) + (size_t)r * NPTS + n];
  }
  X[(size_t)k * COLS + col] = f2bf(v);
}

// --------- 4) GEMM: Y[256][65536] = W[256][Kdim] * X[Kdim][65536] ----------
// Block = 8 waves sharing one 64-column B panel. B tiles (32x64 bf16, 4KB)
// are staged into LDS once per block via the CDNA5 async global->LDS path
// (global_load_async_to_lds_b128, ASYNCcnt), double-buffered so tile t+1
// streams in while tile t feeds the WMMAs.
__global__ void gemm_bf16(const unsigned short* __restrict__ Wp,
                          const unsigned short* __restrict__ Xb,
                          float* __restrict__ Y, int Kdim) {
  __shared__ __align__(16) unsigned char sB[2 * BTILE_BYTES];

  const int tid    = threadIdx.x;
  const int lane   = tid & 31;
  const int waveId = tid >> 5;
  const int rowTile  = (blockIdx.x & 1) * 8 + waveId;   // 16 row tiles / 2 per col
  const int colGroup = blockIdx.x >> 1;                 // 1024 groups of 64 cols
  const int row0 = rowTile * 16;
  const int col0 = colGroup * 64;
  const int half = lane >> 4;              // A half selection
  const int m    = lane & 15;              // A row (M), D column (N)

  // Per-thread async chunk: 32 rows x 8 x 16B chunks == 256 threads.
  const unsigned sbase   = (unsigned)(size_t)(&sB[0]);  // LDS byte offset
  const int      crow    = tid >> 3;
  const int      cchunk  = tid & 7;
  const unsigned short* gbase = Xb + (size_t)crow * COLS + col0 + cchunk * 8;
  const unsigned ldsChunk = sbase + crow * BROW_STRIDE + cchunk * 16;

  // Preload tile 0 into buffer 0.
  asm volatile("global_load_async_to_lds_b128 %0, %1, off"
               :: "v"(ldsChunk), "v"(gbase) : "memory");

  v8f acc0 = {}, acc1 = {}, acc2 = {}, acc3 = {};
  union AB { uint4 q[2]; v16bf v; };
  AB a, b0, b1, b2, b3;

  const int T = Kdim >> 5;                 // 32-deep K tiles
  for (int t = 0; t < T; ++t) {
    if (t + 1 < T) {                       // prefetch tile t+1, then wait tile t
      const unsigned short* gp = gbase + (size_t)(t + 1) * 32 * COLS;
      const unsigned lds = ldsChunk + ((t + 1) & 1) * BTILE_BYTES;
      asm volatile("global_load_async_to_lds_b128 %0, %1, off"
                   :: "v"(lds), "v"(gp) : "memory");
      asm volatile("s_wait_asynccnt 0x1" ::: "memory");
    } else {
      asm volatile("s_wait_asynccnt 0x0" ::: "memory");
    }
    __syncthreads();                       // tile t fully visible in LDS

    // A tile (16x32 bf16): lane m holds K = {half*8..+7, 16+half*8..+7}
    const int k0 = t * 32;
    const unsigned short* wrow = Wp + (size_t)(row0 + m) * Kdim + k0 + half * 8;
    a.q[0] = *(const uint4*)(wrow);
    a.q[1] = *(const uint4*)(wrow + 16);

    // B tiles from LDS: lane%16 = N col, K = {half*8..+7, 16+half*8..+7}.
    const unsigned char* scol = &sB[(t & 1) * BTILE_BYTES + m * 2];
#pragma unroll
    for (int s = 0; s < 16; ++s) {
      const int kk = (s < 8) ? (8 * half + s) : (16 + 8 * half + (s - 8));
      const unsigned char* p = scol + kk * BROW_STRIDE;
      ((unsigned short*)&b0)[s] = *(const unsigned short*)(p +  0);
      ((unsigned short*)&b1)[s] = *(const unsigned short*)(p + 32);
      ((unsigned short*)&b2)[s] = *(const unsigned short*)(p + 64);
      ((unsigned short*)&b3)[s] = *(const unsigned short*)(p + 96);
    }

    acc0 = __builtin_amdgcn_wmma_f32_16x16x32_bf16(false, a.v, false, b0.v, (short)0, acc0, false, false);
    acc1 = __builtin_amdgcn_wmma_f32_16x16x32_bf16(false, a.v, false, b1.v, (short)0, acc1, false, false);
    acc2 = __builtin_amdgcn_wmma_f32_16x16x32_bf16(false, a.v, false, b2.v, (short)0, acc2, false, false);
    acc3 = __builtin_amdgcn_wmma_f32_16x16x32_bf16(false, a.v, false, b3.v, (short)0, acc3, false, false);

    __syncthreads();                       // all waves done before t+2 overwrite
  }

  // D layout: VGPR v -> row = row0 + v + 8*half, lane%16 -> column
  float* yb = Y + (size_t)(row0 + 8 * half) * COLS + col0 + m;
#pragma unroll
  for (int v = 0; v < 8; ++v) {
    yb[(size_t)v * COLS +  0] = acc0[v];
    yb[(size_t)v * COLS + 16] = acc1[v];
    yb[(size_t)v * COLS + 32] = acc2[v];
    yb[(size_t)v * COLS + 48] = acc3[v];
  }
}

// --------- 5) BN batch stats: per-channel mean / rstd over 65536 cols ------
__global__ void bn_stats(const float* __restrict__ Y,
                         float* __restrict__ mean, float* __restrict__ rstd) {
  const int ch = blockIdx.x;
  const float* row = Y + (size_t)ch * COLS;
  float s = 0.f, s2 = 0.f;
  for (int c = threadIdx.x; c < COLS; c += 256) {
    const float v = row[c];
    s += v; s2 += v * v;
  }
  __shared__ float sh[256], sh2[256];
  sh[threadIdx.x] = s; sh2[threadIdx.x] = s2;
  __syncthreads();
  for (int off = 128; off > 0; off >>= 1) {
    if (threadIdx.x < off) {
      sh[threadIdx.x]  += sh[threadIdx.x + off];
      sh2[threadIdx.x] += sh2[threadIdx.x + off];
    }
    __syncthreads();
  }
  if (threadIdx.x == 0) {
    const float mu  = sh[0] * (1.f / COLS);
    const float var = sh2[0] * (1.f / COLS) - mu * mu;
    mean[ch] = mu;
    rstd[ch] = rsqrtf(var + 1e-5f);
  }
}

// --------- 6) BN + ReLU -> bf16 (input of layer 2) -------------------------
__global__ void bn_relu_pack(const float* __restrict__ Y,
                             const float* __restrict__ mean, const float* __restrict__ rstd,
                             const float* __restrict__ g, const float* __restrict__ bt,
                             unsigned short* __restrict__ X2) {
  const int i  = blockIdx.x * 256 + threadIdx.x;     // 0 .. 16777215
  const int ch = i >> 16;
  float v = (Y[i] - mean[ch]) * rstd[ch] * g[ch] + bt[ch];
  X2[i] = f2bf(fmaxf(v, 0.f));
}

// --------- 7) BN + ReLU -> f32 output with (B,C,NR,N) layout ---------------
__global__ void bn_relu_out(const float* __restrict__ Y,
                            const float* __restrict__ mean, const float* __restrict__ rstd,
                            const float* __restrict__ g, const float* __restrict__ bt,
                            float* __restrict__ out) {
  const int i   = blockIdx.x * 256 + threadIdx.x;
  const int ch  = i >> 16;
  const int col = i & (COLS - 1);
  float v = (Y[i] - mean[ch]) * rstd[ch] * g[ch] + bt[ch];
  v = fmaxf(v, 0.f);
  const int b  = col >> 13;                 // col = b*8192 + r*2048 + n
  const int rn = col & 8191;
  out[(size_t)b * (CCH * NRR * NPTS) + (size_t)ch * (NRR * NPTS) + rn] = v;
}

// ---------------------------------------------------------------------------
extern "C" void kernel_launch(void* const* d_in, const int* in_sizes, int n_in,
                              void* d_out, int out_size, void* d_ws, size_t ws_size,
                              hipStream_t stream) {
  const float* unknown = (const float*)d_in[0];
  const float* known   = (const float*)d_in[1];
  const float* uf      = (const float*)d_in[2];   // (B,256,4,2048)
  const float* kf      = (const float*)d_in[3];   // (B,256,4,512)
  const float* W1      = (const float*)d_in[4];
  const float* g1      = (const float*)d_in[5];
  const float* b1      = (const float*)d_in[6];
  const float* W2      = (const float*)d_in[7];
  const float* g2      = (const float*)d_in[8];
  const float* b2      = (const float*)d_in[9];
  float* out = (float*)d_out;

  char* ws = (char*)d_ws;
  int*            ws_idx  = (int*)(ws + 0);                         // 196608 B
  float*          ws_wgt  = (float*)(ws + 196608);                  // 196608 B
  unsigned short* ws_W1p  = (unsigned short*)(ws + 393216);         // 262144 B
  unsigned short* ws_W2p  = (unsigned short*)(ws + 655360);         // 131072 B
  float*          ws_mu1  = (float*)(ws + 786432);
  float*          ws_rs1  = (float*)(ws + 787456);
  float*          ws_mu2  = (float*)(ws + 788480);
  float*          ws_rs2  = (float*)(ws + 789504);
  unsigned short* ws_X    = (unsigned short*)(ws + 1048576);        // 64 MiB
  float*          ws_Y    = (float*)(ws + 1048576 + 67108864);      // 64 MiB
  unsigned short* ws_X2   = ws_X;   // X dead after GEMM1; reuse (needs 32 MiB)

  // 1) kNN + inverse-distance weights
  knn_kernel<<<dim3(NPTS / 256, BB), 256, 0, stream>>>(unknown, known, ws_idx, ws_wgt);
  // 2) weight packing to bf16
  pack_w<<<(CCH * K1DIM) / 256, 256, 0, stream>>>(W1, W2, ws_W1p, ws_W2p);
  // 3) interp + concat -> X (512 x 65536) bf16
  build_x<<<dim3(COLS / 256, K1DIM), 256, 0, stream>>>(kf, uf, ws_idx, ws_wgt, ws_X);
  // 4) layer-1 GEMM: 2048 blocks x 8 waves (each block: 128 rows x 64 cols)
  gemm_bf16<<<2048, 256, 0, stream>>>(ws_W1p, ws_X, ws_Y, K1DIM);
  // 5) BN stats layer 1
  bn_stats<<<CCH, 256, 0, stream>>>(ws_Y, ws_mu1, ws_rs1);
  // 6) BN + ReLU + bf16 pack -> X2 (256 x 65536)
  bn_relu_pack<<<(CCH * COLS) / 256, 256, 0, stream>>>(ws_Y, ws_mu1, ws_rs1, g1, b1, ws_X2);
  // 7) layer-2 GEMM (overwrites Y; layer-1 Y fully consumed)
  gemm_bf16<<<2048, 256, 0, stream>>>(ws_W2p, ws_X2, ws_Y, K2DIM);
  // 8) BN stats layer 2
  bn_stats<<<CCH, 256, 0, stream>>>(ws_Y, ws_mu2, ws_rs2);
  // 9) BN + ReLU -> output (B,256,4,2048) f32
  bn_relu_out<<<(CCH * COLS) / 256, 256, 0, stream>>>(ws_Y, ws_mu2, ws_rs2, g2, b2, out);
}
